// conv_matrix_layer_68186900791733
// MI455X (gfx1250) — matrix operand order repaired and checked by reading, not compiled, not run on hardware
//
#include <hip/hip_runtime.h>

typedef _Float16 v16h __attribute__((ext_vector_type(16)));
typedef float    v8f  __attribute__((ext_vector_type(8)));
typedef __fp16   h2   __attribute__((ext_vector_type(2)));   // matches cvt_pkrtz return
typedef _Float16 h4   __attribute__((ext_vector_type(4)));
typedef float    f4   __attribute__((ext_vector_type(4)));

namespace {
constexpr int   kB        = 4096;
constexpr int   kM        = 128;
constexpr int   kK        = kM * kM;          // 16384
constexpr int   kNMom     = 255;              // t^0 .. t^254
constexpr int   kMomBlk   = 512;
constexpr int   kSplit    = 4;
constexpr int   kJChunk   = kM / kSplit;      // 32
constexpr float kEps      = 1e-5f;
constexpr float kInvBM    = 1.0f / 524288.0f;

// workspace layout (bytes), all 256B aligned
constexpr size_t OFF_TPART = 0;                                        // 255*512*4 = 522240
constexpr size_t OFF_T     = 522496;                                   // 255 floats
constexpr size_t OFF_SCALE = OFF_T + 1024;
constexpr size_t OFF_SHIFT = OFF_SCALE + 512;
constexpr size_t OFF_CB    = OFF_SHIFT + 512;
constexpr size_t OFF_W2    = OFF_CB + 512;                             // 128*16384 f16 = 4MB
constexpr size_t OFF_PART  = OFF_W2 + (size_t)kM * kK * 2;             // 4*4096*128 f32 = 8MB
} // namespace

__device__ __forceinline__ float wave_reduce(float s) {
#pragma unroll
  for (int off = 16; off > 0; off >>= 1) s += __shfl_xor(s, off, 32);
  return s;
}

// valid in thread 0 only; callable once per kernel
__device__ __forceinline__ float block_reduce_256(float s) {
  __shared__ float red[8];
  const int lane = threadIdx.x & 31, wid = threadIdx.x >> 5;
  s = wave_reduce(s);
  if (lane == 0) red[wid] = s;
  __syncthreads();
  float tot = 0.0f;
  if (threadIdx.x == 0) {
#pragma unroll
    for (int w = 0; w < 8; ++w) tot += red[w];
  }
  return tot;
}

// ---- K1: per-block partial power sums T_m = sum t^m, m = 0..254 -------------
__global__ void k_moments(const float* __restrict__ X, float* __restrict__ Tpart) {
  __shared__ float wl[8][kNMom];
  const int tid = threadIdx.x, lane = tid & 31, wid = tid >> 5;
  const size_t base = (size_t)blockIdx.x * 1024 + (size_t)tid * 4;
  f4 x = *(const f4*)(X + base);
  float t0 = x[0] * x[0], t1 = x[1] * x[1], t2 = x[2] * x[2], t3 = x[3] * x[3];
  float p0 = 1.0f, p1 = 1.0f, p2 = 1.0f, p3 = 1.0f;
  for (int m = 0; m < kNMom; ++m) {
    float s = (p0 + p1) + (p2 + p3);
    s = wave_reduce(s);
    if (lane == 0) wl[wid][m] = s;
    p0 *= t0; p1 *= t1; p2 *= t2; p3 *= t3;
  }
  __syncthreads();
  for (int m = tid; m < kNMom; m += blockDim.x) {
    float s = 0.0f;
#pragma unroll
    for (int w = 0; w < 8; ++w) s += wl[w][m];
    Tpart[(size_t)m * kMomBlk + blockIdx.x] = s;
  }
}

// ---- K2: finalize moments ---------------------------------------------------
__global__ void k_moments_final(const float* __restrict__ Tpart, float* __restrict__ T) {
  const int m = blockIdx.x;
  float s = Tpart[(size_t)m * kMomBlk + threadIdx.x] +
            Tpart[(size_t)m * kMomBlk + threadIdx.x + 256];
  float tot = block_reduce_256(s);
  if (threadIdx.x == 0) T[m] = tot;
}

// ---- K3: per-channel BN scale/shift ----------------------------------------
__global__ void k_stats(const float* __restrict__ T, const float* __restrict__ gamma,
                        const float* __restrict__ beta, float* __restrict__ scale,
                        float* __restrict__ shift) {
  const int j   = threadIdx.x;
  const float mean = T[j] * kInvBM;
  const float ex2  = T[2 * j] * kInvBM;
  const float var  = fmaxf(ex2 - mean * mean, 0.0f);
  const float sc   = gamma[j] * rsqrtf(var + kEps);
  scale[j] = sc;
  shift[j] = beta[j] - sc * mean;
}

// ---- K4: W2 = f16(conv_w * scale_j) ----------------------------------------
__global__ void k_wprep(const float* __restrict__ conv_w, const float* __restrict__ scale,
                        _Float16* __restrict__ W2) {
  const size_t f = ((size_t)blockIdx.x * blockDim.x + threadIdx.x) * 4;
  const int j = (int)((f >> 7) & 127);
  const float sc = scale[j];
  f4 w = *(const f4*)(conv_w + f);
  h4 o;
  o[0] = (_Float16)(w[0] * sc); o[1] = (_Float16)(w[1] * sc);
  o[2] = (_Float16)(w[2] * sc); o[3] = (_Float16)(w[3] * sc);
  *(h4*)(W2 + f) = o;
}

// ---- K5: c[o] = conv_b[o] + sum_k conv_w[o,k] * shift[k/128] ---------------
__global__ void k_bias(const float* __restrict__ conv_w, const float* __restrict__ conv_b,
                       const float* __restrict__ shift, float* __restrict__ cbias) {
  const int o = blockIdx.x;
  const float* row = conv_w + (size_t)o * kK;
  float s = 0.0f;
  for (int k = threadIdx.x; k < kK; k += 256) s += row[k] * shift[k >> 7];
  float tot = block_reduce_256(s);
  if (threadIdx.x == 0) cbias[o] = conv_b[o] + tot;
}

// ---- K6: WMMA GEMM, A generated on the fly as running powers t^j -----------
// block = 128 threads (4 waves); wave -> 16 rows x 128 cols; grid (64, kSplit)
__launch_bounds__(128, 1)
__global__ void k_gemm(const float* __restrict__ X, const _Float16* __restrict__ W2,
                       float* __restrict__ part) {
  const int lane = threadIdx.x & 31;
  const int wave = threadIdx.x >> 5;
  const int m    = lane & 15;          // A row / B col within 16-lane group
  const int hi   = lane >> 4;          // upper half-wave flag
  const int kb   = hi * 8;             // A-fragment K base split (ISA layout)
  const int rowbase = blockIdx.x * 64 + wave * 16;
  const int ky   = blockIdx.y;
  const int j0   = ky * kJChunk;

  // load t = x^2 for this lane's 64 (row, i) slots, laid out exactly as the
  // A fragments need them: tt[s*16 + e]  <->  i = 32 s + kb + 16*(e>=8) + (e&7)
  const float* Xrow = X + (size_t)(rowbase + m) * kM;
  float tt[64];
#pragma unroll
  for (int s = 0; s < 4; ++s) {
#pragma unroll
    for (int g = 0; g < 2; ++g) {
      const int i0 = 32 * s + kb + 16 * g;
      f4 a = *(const f4*)(Xrow + i0);
      f4 b = *(const f4*)(Xrow + i0 + 4);
      const int t0 = s * 16 + g * 8;
      tt[t0 + 0] = a[0] * a[0]; tt[t0 + 1] = a[1] * a[1];
      tt[t0 + 2] = a[2] * a[2]; tt[t0 + 3] = a[3] * a[3];
      tt[t0 + 4] = b[0] * b[0]; tt[t0 + 5] = b[1] * b[1];
      tt[t0 + 6] = b[2] * b[2]; tt[t0 + 7] = b[3] * b[3];
    }
  }

  // P = t^(32*ky) via 5 squarings (uniform branches)
  float P[64];
#pragma unroll
  for (int r = 0; r < 64; ++r) {
    float u = tt[r];
    u *= u; u *= u; u *= u; u *= u; u *= u;     // t^32
    float p = 1.0f;
    if (ky & 1) p *= u;
    float u2 = u * u;
    if (ky & 2) p *= u2;
    P[r] = p;
  }

  v8f acc[8];
#pragma unroll
  for (int c = 0; c < 8; ++c)
#pragma unroll
    for (int e = 0; e < 8; ++e) acc[c][e] = 0.0f;

  // per-lane B base: W2[(c*16+m)*16384 + j*128 + 32*s + 8*hi], halves 0-7 and 16-23
  const _Float16* wbase = W2 + (size_t)m * kK + 8 * hi;

#pragma unroll 1
  for (int j = j0; j < j0 + kJChunk; ++j) {
#pragma unroll
    for (int s = 0; s < 4; ++s) {
      union AFrag { v16h v; h2 p[8]; } af;
#pragma unroll
      for (int q = 0; q < 8; ++q)
        af.p[q] = __builtin_amdgcn_cvt_pkrtz(P[s * 16 + 2 * q], P[s * 16 + 2 * q + 1]);
#pragma unroll
      for (int c = 0; c < 8; ++c) {
        const f4* bp = (const f4*)(wbase + (size_t)c * 16 * kK + j * 128 + 32 * s);
        union BFrag { v16h v; f4 f[2]; } bf;
        bf.f[0] = bp[0];
        bf.f[1] = bp[2];
        acc[c] = __builtin_amdgcn_wmma_f32_16x16x32_f16(
            false, af.v, false, bf.v, (short)0, acc[c], false, false);
      }
    }
#pragma unroll
    for (int r = 0; r < 64; ++r) P[r] *= tt[r];
  }

  // D layout: reg r -> row M = r + 8*hi, col N = lane%16
  float* out = part + (size_t)ky * kB * kM;
#pragma unroll
  for (int c = 0; c < 8; ++c) {
    const int col = c * 16 + m;
#pragma unroll
    for (int r = 0; r < 8; ++r) {
      const int row = rowbase + r + 8 * hi;
      out[(size_t)row * kM + col] = acc[c][r];
    }
  }
}

// ---- K7: sum the kSplit partials and add bias ------------------------------
__global__ void k_combine(const float* __restrict__ part, const float* __restrict__ cbias,
                          float* __restrict__ out) {
  const size_t f = ((size_t)blockIdx.x * blockDim.x + threadIdx.x) * 4;
  f4 p0 = *(const f4*)(part + f);
  f4 p1 = *(const f4*)(part + (size_t)1 * kB * kM + f);
  f4 p2 = *(const f4*)(part + (size_t)2 * kB * kM + f);
  f4 p3 = *(const f4*)(part + (size_t)3 * kB * kM + f);
  f4 cv = *(const f4*)(cbias + (f & 127));
  f4 o;
#pragma unroll
  for (int e = 0; e < 4; ++e) o[e] = ((p0[e] + p1[e]) + (p2[e] + p3[e])) + cv[e];
  *(f4*)(out + f) = o;
}

extern "C" void kernel_launch(void* const* d_in, const int* in_sizes, int n_in,
                              void* d_out, int out_size, void* d_ws, size_t ws_size,
                              hipStream_t stream) {
  (void)in_sizes; (void)n_in; (void)out_size; (void)ws_size;
  const float* X      = (const float*)d_in[0];
  const float* gamma  = (const float*)d_in[1];
  const float* beta   = (const float*)d_in[2];
  const float* conv_w = (const float*)d_in[3];
  const float* conv_b = (const float*)d_in[4];
  float* out = (float*)d_out;

  char* ws = (char*)d_ws;
  float*     Tpart = (float*)(ws + OFF_TPART);
  float*     T     = (float*)(ws + OFF_T);
  float*     scale = (float*)(ws + OFF_SCALE);
  float*     shift = (float*)(ws + OFF_SHIFT);
  float*     cbias = (float*)(ws + OFF_CB);
  _Float16*  W2    = (_Float16*)(ws + OFF_W2);
  float*     part  = (float*)(ws + OFF_PART);

  k_moments<<<kMomBlk, 256, 0, stream>>>(X, Tpart);
  k_moments_final<<<kNMom, 256, 0, stream>>>(Tpart, T);
  k_stats<<<1, kM, 0, stream>>>(T, gamma, beta, scale, shift);
  k_wprep<<<(kM * kK) / (256 * 4), 256, 0, stream>>>(conv_w, scale, W2);
  k_bias<<<kM, 256, 0, stream>>>(conv_w, conv_b, shift, cbias);
  k_gemm<<<dim3(kB / 64, kSplit), 128, 0, stream>>>(X, W2, part);
  k_combine<<<(kB * kM) / (256 * 4), 256, 0, stream>>>(part, cbias, out);
}
